// Quantizer_36335423324704
// MI455X (gfx1250) — matrix operand order repaired and checked by reading, not compiled, not run on hardware
//
#include <hip/hip_runtime.h>
#include <stdint.h>

// ---------------------------------------------------------------------------
// VQ-VAE quantizer forward for MI455X (gfx1250, wave32, WMMA).
// Phase 1: argmin_k(||w_k||^2 - 2 z.w_k) via v_wmma_f32_16x16x32_bf16 with
//          hi/lo bf16 split of both operands (near-fp32 accurate scores).
// Phase 2: gather + [B,H,W,D]->[B,D,H,W] transpose via LDS + loss reduction.
// ---------------------------------------------------------------------------

typedef __attribute__((ext_vector_type(16))) __bf16          v16bf;
typedef __attribute__((ext_vector_type(8)))  float           v8f;
typedef __attribute__((ext_vector_type(8)))  unsigned short  u16x8;
typedef __attribute__((ext_vector_type(16))) unsigned short  u16x16;

#define D_      64
#define K_      1024
#define HALF_K  512
#define B_      32
#define H_      64
#define W_      64
#define NROWS   (B_ * H_ * W_)            // 131072 vectors
#define RTILES  (NROWS / 16)              // 8192 16-row tiles
#define P1_GRID 256
#define P1_BLK  256
#define WPB     (P1_BLK / 32)             // 8 waves / block
#define RT_PW   (RTILES / (P1_GRID * WPB))// 4 row tiles per wave
#define LDS_BYTES (2 * HALF_K * D_ * 2 + HALF_K * 4)  // hi+lo bf16 + wsq = 133120

static __device__ __forceinline__ unsigned short f32_to_bf16_rn(float f) {
  unsigned int u = __builtin_bit_cast(unsigned int, f);
  u += 0x7fffu + ((u >> 16) & 1u);        // round-to-nearest-even
  return (unsigned short)(u >> 16);
}
static __device__ __forceinline__ float bf16f(unsigned short h) {
  unsigned int u = ((unsigned int)h) << 16;
  return __builtin_bit_cast(float, u);
}

__global__ __launch_bounds__(P1_BLK, 1)
void vq_argmin_wmma(const float* __restrict__ z, const float* __restrict__ wgt,
                    float* __restrict__ out_idx) {
  extern __shared__ unsigned char smem[];
  unsigned short* whi = (unsigned short*)smem;               // [512][64] bf16 hi
  unsigned short* wlo = whi + HALF_K * D_;                   // [512][64] bf16 lo
  float* wsqh = (float*)(smem + (size_t)2 * HALF_K * D_ * 2);// [512] ||w||^2 (f32)

  const int tid  = threadIdx.x;
  const int wave = tid >> 5;
  const int lane = tid & 31;
  const int col  = lane & 15;          // M (A rows) / N (B,C columns)
  const int kbA  = (lane >> 4) * 8;    // A-fragment K sub-offset
  const int kbB  = (lane >> 4) * 8;    // B-fragment K sub-offset
  const int rtBase = (blockIdx.x * WPB + wave) * RT_PW;

  float minv[RT_PW][8];
  int   mini[RT_PW][8];
  #pragma unroll
  for (int it = 0; it < RT_PW; ++it)
    #pragma unroll
    for (int r = 0; r < 8; ++r) { minv[it][r] = 3.402823e38f; mini[it][r] = 0; }

  for (int half = 0; half < 2; ++half) {
    __syncthreads();
    // Cooperative stage of 512 codes: fp32 -> bf16 hi/lo split + ||w||^2.
    #pragma unroll
    for (int cc = 0; cc < HALF_K / P1_BLK; ++cc) {
      const int cl = tid + cc * P1_BLK;                  // local code 0..511
      const float* wr = wgt + (size_t)(half * HALF_K + cl) * D_;
      unsigned short* hr = whi + cl * D_;
      unsigned short* lr = wlo + cl * D_;
      float s = 0.f;
      #pragma unroll
      for (int d = 0; d < D_; d += 4) {
        float4 v4 = *(const float4*)(wr + d);
        float xs[4] = {v4.x, v4.y, v4.z, v4.w};
        #pragma unroll
        for (int j = 0; j < 4; ++j) {
          float x = xs[j];
          s = fmaf(x, x, s);
          unsigned short hb = f32_to_bf16_rn(x);
          hr[d + j] = hb;
          lr[d + j] = f32_to_bf16_rn(x - bf16f(hb));
        }
      }
      wsqh[cl] = s;
    }
    __syncthreads();

    #pragma unroll
    for (int it = 0; it < RT_PW; ++it) {
      const int r0 = (rtBase + it) * 16;
      const float* zr = z + (size_t)(r0 + col) * D_;
      // A fragments (16x32 bf16 per K-chunk), hi/lo split, ISA lane layout.
      v16bf ahi[2], alo[2];
      #pragma unroll
      for (int c = 0; c < 2; ++c) {
        float f[16];
        float4 a0 = *(const float4*)(zr + c * 32 + kbA);
        float4 a1 = *(const float4*)(zr + c * 32 + kbA + 4);
        float4 a2 = *(const float4*)(zr + c * 32 + kbA + 16);
        float4 a3 = *(const float4*)(zr + c * 32 + kbA + 20);
        f[0]=a0.x; f[1]=a0.y; f[2]=a0.z;  f[3]=a0.w;
        f[4]=a1.x; f[5]=a1.y; f[6]=a1.z;  f[7]=a1.w;
        f[8]=a2.x; f[9]=a2.y; f[10]=a2.z; f[11]=a2.w;
        f[12]=a3.x;f[13]=a3.y;f[14]=a3.z; f[15]=a3.w;
        u16x16 hv, lv;
        #pragma unroll
        for (int e = 0; e < 16; ++e) {
          unsigned short hb = f32_to_bf16_rn(f[e]);
          hv[e] = hb;
          lv[e] = f32_to_bf16_rn(f[e] - bf16f(hb));
        }
        ahi[c] = __builtin_bit_cast(v16bf, hv);
        alo[c] = __builtin_bit_cast(v16bf, lv);
      }

      #pragma unroll 2
      for (int t = 0; t < HALF_K / 16; ++t) {            // 32 code tiles / half
        v8f acc = {0.f, 0.f, 0.f, 0.f, 0.f, 0.f, 0.f, 0.f};
        #pragma unroll
        for (int c = 0; c < 2; ++c) {
          const unsigned short* bph = whi + (t * 16 + col) * D_ + c * 32 + kbB;
          const unsigned short* bpl = wlo + (t * 16 + col) * D_ + c * 32 + kbB;
          u16x8 h0 = *(const u16x8*)bph;
          u16x8 h1 = *(const u16x8*)(bph + 16);
          u16x8 l0 = *(const u16x8*)bpl;
          u16x8 l1 = *(const u16x8*)(bpl + 16);
          v16bf bh = __builtin_bit_cast(v16bf,
              __builtin_shufflevector(h0, h1, 0,1,2,3,4,5,6,7,8,9,10,11,12,13,14,15));
          v16bf bl = __builtin_bit_cast(v16bf,
              __builtin_shufflevector(l0, l1, 0,1,2,3,4,5,6,7,8,9,10,11,12,13,14,15));
          // S += Ahi*Bhi + Ahi*Blo + Alo*Bhi  (fp32-accurate z.w^T)
          acc = __builtin_amdgcn_wmma_f32_16x16x32_bf16(false, ahi[c], false, bh,
                                                        (short)0, acc, false, false);
          acc = __builtin_amdgcn_wmma_f32_16x16x32_bf16(false, ahi[c], false, bl,
                                                        (short)0, acc, false, false);
          acc = __builtin_amdgcn_wmma_f32_16x16x32_bf16(false, alo[c], false, bh,
                                                        (short)0, acc, false, false);
        }
        const float wq = wsqh[t * 16 + col];
        const int   cg = half * HALF_K + t * 16 + col;
        #pragma unroll
        for (int r = 0; r < 8; ++r) {
          float dist = fmaf(-2.f, acc[r], wq);           // ||w||^2 - 2 z.w
          if (dist < minv[it][r]) { minv[it][r] = dist; mini[it][r] = cg; }
        }
      }
    }
  }

  // Cross-lane argmin within each 16-lane half; ties -> lowest index
  // (matches jnp.argmin first-occurrence semantics).
  #pragma unroll
  for (int it = 0; it < RT_PW; ++it) {
    #pragma unroll
    for (int off = 8; off >= 1; off >>= 1) {
      #pragma unroll
      for (int r = 0; r < 8; ++r) {
        float ov = __shfl_xor(minv[it][r], off, 32);
        int   oi = __shfl_xor(mini[it][r], off, 32);
        if (ov < minv[it][r] || (ov == minv[it][r] && oi < mini[it][r])) {
          minv[it][r] = ov; mini[it][r] = oi;
        }
      }
    }
    const int r0 = (rtBase + it) * 16;
    if (lane == 0) {                       // rows M = 0..7 of the tile
      #pragma unroll
      for (int r = 0; r < 8; ++r) out_idx[r0 + r] = (float)mini[it][r];
    } else if (lane == 16) {               // rows M = 8..15 of the tile
      #pragma unroll
      for (int r = 0; r < 8; ++r) out_idx[r0 + 8 + r] = (float)mini[it][r];
    }
  }
}

// ---------------------------------------------------------------------------
// Phase 2: gather codebook rows, transpose via LDS, accumulate loss.
// One block per (b,h): 64 w-positions x 64 dims.
// ---------------------------------------------------------------------------
__global__ __launch_bounds__(256, 2)
void vq_gather_loss(const float* __restrict__ z, const float* __restrict__ wgt,
                    const float* __restrict__ idx_f, float* __restrict__ out_q,
                    float* __restrict__ loss) {
  __shared__ float qs[W_][D_ + 1];   // +1 pad -> conflict-free transpose
  __shared__ int   idxs[W_];
  __shared__ float red[8];

  const int b   = blockIdx.x >> 6;
  const int h   = blockIdx.x & 63;
  const int tid = threadIdx.x;

  if (tid < W_) idxs[tid] = (int)idx_f[(b * H_ + h) * W_ + tid];
  __syncthreads();

  const float* zrow = z + (size_t)((b * H_ + h) * W_) * D_;  // [w][d], d fastest
  float lsum = 0.f;
  #pragma unroll
  for (int i = 0; i < (W_ * D_) / 256; ++i) {   // 16 iters, coalesced on z
    const int e = tid + i * 256;
    const int w = e >> 6, d = e & 63;
    float q  = wgt[(size_t)idxs[w] * D_ + d];
    float df = q - zrow[e];
    lsum = fmaf(df, df, lsum);
    qs[w][d] = q;
  }
  __syncthreads();

  // out[b][d][h][w]: 64 contiguous w per (d) row -> coalesced stores.
  float* ob = out_q + (size_t)b * D_ * H_ * W_ + (size_t)h * W_;
  #pragma unroll
  for (int i = 0; i < (W_ * D_) / 256; ++i) {
    const int e = tid + i * 256;
    const int d = e >> 6, w = e & 63;
    ob[(size_t)d * H_ * W_ + w] = qs[w][d];
  }

  // loss = (QUANT_F + COMMIT_F) * mean((q - z)^2) = 1.25 * sum / (N*D)
  #pragma unroll
  for (int off = 16; off >= 1; off >>= 1) lsum += __shfl_xor(lsum, off, 32);
  if ((tid & 31) == 0) red[tid >> 5] = lsum;
  __syncthreads();
  if (tid == 0) {
    float s = 0.f;
    #pragma unroll
    for (int i = 0; i < 8; ++i) s += red[i];
    atomicAdd(loss, s * (1.25f / (float)((size_t)NROWS * D_)));
  }
}

__global__ void vq_zero_loss(float* loss) { *loss = 0.f; }

extern "C" void kernel_launch(void* const* d_in, const int* in_sizes, int n_in,
                              void* d_out, int out_size, void* d_ws, size_t ws_size,
                              hipStream_t stream) {
  (void)in_sizes; (void)n_in; (void)out_size; (void)d_ws; (void)ws_size;
  const float* z   = (const float*)d_in[0];   // [32,64,64,64] fp32
  const float* wgt = (const float*)d_in[1];   // [1024,64] fp32

  float* out      = (float*)d_out;
  float* out_qv   = out;                                   // [32,64,64,64] (B,D,H,W)
  float* out_idx  = out + (size_t)B_ * D_ * H_ * W_;       // [32,1,64,64] as float
  float* out_loss = out_idx + (size_t)B_ * H_ * W_;        // scalar

  (void)hipFuncSetAttribute(reinterpret_cast<const void*>(vq_argmin_wmma),
                            hipFuncAttributeMaxDynamicSharedMemorySize, LDS_BYTES);

  vq_zero_loss<<<1, 1, 0, stream>>>(out_loss);
  vq_argmin_wmma<<<dim3(P1_GRID), dim3(P1_BLK), LDS_BYTES, stream>>>(z, wgt, out_idx);
  vq_gather_loss<<<dim3(B_ * H_), dim3(256), 0, stream>>>(z, wgt, out_idx, out_qv, out_loss);
}
